// NodePositionHead_29652454212055
// MI455X (gfx1250) — matrix operand order repaired and checked by reading, not compiled, not run on hardware
//
#include <hip/hip_runtime.h>
#include <cstdint>

typedef __attribute__((ext_vector_type(16))) _Float16 v16h;
typedef __attribute__((ext_vector_type(8)))  float    v8f;

union Frag {
    v16h v;
    _Float16 h[16];
    uint32_t u[8];
};

#define NB 32
#define NS 256
#define NE 768
#define NH 32
#define ND 24
#define DP 32          // padded head dim
#define SB 64          // s-rows per attention workgroup
#define SCALING 0.2041241452319315f   // 24^-0.5

#define AS_STRIDE 17   // As row stride in dwords (banked)
#define BS_STRIDE 132  // Bs row stride in dwords (banked)

__device__ __forceinline__ uint32_t pkf16(float x, float y) {
    union { _Float16 h[2]; uint32_t u; } t;
    t.h[0] = (_Float16)x;
    t.h[1] = (_Float16)y;
    return t.u;
}

// ---------------------------------------------------------------------------
// Kernel A: QKV projection.  X(8192x768) @ {Wq,Wk,Wv}(768x768) + bias.
// Block = 8 waves -> 64x128 output tile; LDS-staged f16 tiles; each wave
// computes 4 N-tiles reusing one A fragment per 32-wide k-step.
// q -> [b,h,s,32] (scaled), k -> [b,h,t,32], v -> [b,h,d,t] (transposed).
// ---------------------------------------------------------------------------
__global__ __launch_bounds__(256) void qkv_proj_kernel(
    const float* __restrict__ feats,
    const float* __restrict__ Wq, const float* __restrict__ bq,
    const float* __restrict__ Wk, const float* __restrict__ bk,
    const float* __restrict__ Wv, const float* __restrict__ bv,
    _Float16* __restrict__ qbuf,   // [B,H,S,DP]
    _Float16* __restrict__ kbuf,   // [B,H,S,DP]
    _Float16* __restrict__ vTbuf)  // [B,H,DP,S]
{
    __shared__ uint32_t As[64 * AS_STRIDE];   // [row][k-pair] f16x2
    __shared__ uint32_t Bs[16 * BS_STRIDE];   // [k-pair][n]   f16x2 (k, k+1)

    const int tid = threadIdx.x;
    const int wv  = tid >> 5;
    const int ln  = tid & 31;
    const int hf  = ln >> 4;
    const int n16 = ln & 15;

    const int n0  = blockIdx.x * 128;   // 6 col blocks
    const int m0  = blockIdx.y * 64;    // 128 row blocks
    const int mat = blockIdx.z;         // 0=q 1=k 2=v (block-uniform)

    const float* W  = (mat == 0) ? Wq : (mat == 1) ? Wk : Wv;
    const float* bp = (mat == 0) ? bq : (mat == 1) ? bk : bv;

    const int mt     = wv >> 1;         // wave's row tile (0..3)
    const int ntbase = (wv & 1) * 4;    // wave's 4 col tiles

    // staging coordinates
    const int arow = tid >> 2;          // 0..63
    const int akc  = (tid & 3) * 8;     // k chunk 0/8/16/24
    const int bkr  = (tid >> 4) * 2;    // W row pair 0,2,..,30
    const int bnc  = (tid & 15) * 8;    // W col chunk

    v8f c[4] = {{}, {}, {}, {}};

    for (int k0 = 0; k0 < NE; k0 += 32) {
        // ---- stage A tile (feats 64x32 -> f16 pairs) ----
        {
            const float4* src = (const float4*)(feats + (size_t)(m0 + arow) * NE + k0 + akc);
            const float4 f0 = src[0];
            const float4 f1 = src[1];
            uint32_t* dst = &As[arow * AS_STRIDE + (akc >> 1)];
            dst[0] = pkf16(f0.x, f0.y);
            dst[1] = pkf16(f0.z, f0.w);
            dst[2] = pkf16(f1.x, f1.y);
            dst[3] = pkf16(f1.z, f1.w);
        }
        // ---- stage B tile (W 32x128 -> packed [k/2][n]) ----
        {
            const float4* r0 = (const float4*)(W + (size_t)(k0 + bkr) * NE + n0 + bnc);
            const float4* r1 = (const float4*)(W + (size_t)(k0 + bkr + 1) * NE + n0 + bnc);
            const float4 a0 = r0[0], a1 = r0[1];
            const float4 b0 = r1[0], b1 = r1[1];
            uint32_t* dst = &Bs[(bkr >> 1) * BS_STRIDE + bnc];
            dst[0] = pkf16(a0.x, b0.x);
            dst[1] = pkf16(a0.y, b0.y);
            dst[2] = pkf16(a0.z, b0.z);
            dst[3] = pkf16(a0.w, b0.w);
            dst[4] = pkf16(a1.x, b1.x);
            dst[5] = pkf16(a1.y, b1.y);
            dst[6] = pkf16(a1.z, b1.z);
            dst[7] = pkf16(a1.w, b1.w);
        }
        __syncthreads();

        // ---- compute: one A fragment, 4 WMMAs ----
        Frag a;
        #pragma unroll
        for (int j = 0; j < 8; ++j) {
            const int koff = hf * 16 + 2 * j;
            a.u[j] = As[(mt * 16 + n16) * AS_STRIDE + (koff >> 1)];
        }
        #pragma unroll
        for (int nt2 = 0; nt2 < 4; ++nt2) {
            Frag b;
            #pragma unroll
            for (int j = 0; j < 8; ++j) {
                b.u[j] = Bs[(hf * 8 + j) * BS_STRIDE + (ntbase + nt2) * 16 + n16];
            }
            c[nt2] = __builtin_amdgcn_wmma_f32_16x16x32_f16(false, a.v, false, b.v,
                                                            (short)0, c[nt2], false, false);
        }
        __syncthreads();
    }

    // ---- write out with bias (+ q scaling), scatter into padded layouts ----
    #pragma unroll
    for (int nt2 = 0; nt2 < 4; ++nt2) {
        const int   col    = n0 + (ntbase + nt2) * 16 + n16;
        const float bias_n = bp[col];
        const int   h      = col / ND;
        const int   d      = col - h * ND;
        #pragma unroll
        for (int i = 0; i < 8; ++i) {
            const int m   = i + hf * 8;        // C: M = i + 8*half, N = lane%16
            const int row = m0 + mt * 16 + m;
            const int bb  = row >> 8;
            const int s   = row & 255;
            float val = c[nt2][i] + bias_n;
            if (mat == 0) {
                val *= SCALING;
                qbuf[(((size_t)bb * NH + h) * NS + s) * DP + d] = (_Float16)val;
            } else if (mat == 1) {
                kbuf[(((size_t)bb * NH + h) * NS + s) * DP + d] = (_Float16)val;
            } else {
                vTbuf[(((size_t)bb * NH + h) * DP + d) * NS + s] = (_Float16)val;
            }
        }
    }
}

// ---------------------------------------------------------------------------
// Kernel B: attention per (b*h, 64-row s-block).
// scores = q kT (WMMA, K=32 covers padded D) + bias -> LDS; softmax;
// per axis x: Px = probs*delta_x (f16 in LDS), attn_x = Px @ vT (WMMA),
// dot with Wf slice, per-head partials to workspace.
// ---------------------------------------------------------------------------
__global__ __launch_bounds__(256) void attn_kernel(
    const float* __restrict__ attn_bias,   // [B,H,S,S]
    const float* __restrict__ delta_pos,   // [B,S,S,3]
    const float* __restrict__ Wfx,
    const float* __restrict__ Wfy,
    const float* __restrict__ Wfz,
    const _Float16* __restrict__ qbuf,     // [B,H,S,DP]
    const _Float16* __restrict__ kbuf,     // [B,H,S,DP]
    const _Float16* __restrict__ vTbuf,    // [B,H,DP,S]
    float* __restrict__ partial)           // [B,S,3,H]
{
    __shared__ float    sc[SB * 260];      // scores / probs, padded stride
    __shared__ uint32_t px[SB * 128];      // f16 Px tile; 'red' aliases this

    const int bh   = blockIdx.x;           // b*H + h
    const int s0   = blockIdx.y * SB;
    const int b    = bh / NH;
    const int h    = bh % NH;
    const int tid  = threadIdx.x;
    const int wv   = tid >> 5;
    const int ln   = tid & 31;
    const int hf   = ln >> 4;
    const int n16  = ln & 15;

    // ---- Phase 1: scores = q @ kT + bias ----
    const int mt = wv >> 1;                // 4 row tiles, 2 waves each
    Frag a;
    {
        const _Float16* qr = qbuf + (((size_t)bh * NS) + s0 + mt * 16 + n16) * DP;
        #pragma unroll
        for (int j = 0; j < 8; ++j) {
            const int koff = hf * 16 + 2 * j;
            a.u[j] = *(const uint32_t*)(qr + koff);
        }
    }
    for (int nn = 0; nn < 8; ++nn) {
        const int nt = (wv & 1) * 8 + nn;
        const int t0 = nt * 16;
        Frag kb;
        #pragma unroll
        for (int j = 0; j < 8; ++j) {
            const int d = hf * 16 + 2 * j;
            kb.u[j] = *(const uint32_t*)(kbuf + (((size_t)bh * NS) + t0 + n16) * DP + d);
        }
        v8f c = {};
        c = __builtin_amdgcn_wmma_f32_16x16x32_f16(false, a.v, false, kb.v,
                                                   (short)0, c, false, false);
        const float* bias = attn_bias + ((size_t)bh * NS + (s0 + mt * 16)) * NS + t0;
        #pragma unroll
        for (int i = 0; i < 8; ++i) {
            const int m = i + hf * 8;
            sc[(mt * 16 + m) * 260 + t0 + n16] = c[i] + bias[(size_t)m * NS + n16];
        }
    }
    __syncthreads();

    // ---- Phase 2: row softmax (wave wv owns rows wv*8 .. wv*8+7) ----
    for (int rr = 0; rr < 8; ++rr) {
        const int r = wv * 8 + rr;
        float v[8];
        float mx = -3.0e38f;
        #pragma unroll
        for (int ii = 0; ii < 8; ++ii) {
            v[ii] = sc[r * 260 + ln + 32 * ii];
            mx = fmaxf(mx, v[ii]);
        }
        for (int off = 16; off; off >>= 1) mx = fmaxf(mx, __shfl_xor(mx, off));
        float sum = 0.f;
        #pragma unroll
        for (int ii = 0; ii < 8; ++ii) { v[ii] = __expf(v[ii] - mx); sum += v[ii]; }
        for (int off = 16; off; off >>= 1) sum += __shfl_xor(sum, off);
        const float inv = 1.f / sum;
        #pragma unroll
        for (int ii = 0; ii < 8; ++ii) sc[r * 260 + ln + 32 * ii] = v[ii] * inv;
    }
    __syncthreads();

    // ---- Phase 3: per axis x ----
    _Float16* pxh = (_Float16*)px;
    float*    red = (float*)px;            // [64][32], aliases px (safe: barriers)
    const int mt2 = wv >> 1;
    const int nt2 = wv & 1;
    for (int x = 0; x < 3; ++x) {
        // Build Px = probs * delta_x as f16 in LDS
        for (int idx = tid; idx < SB * NS; idx += 256) {
            const int s = idx >> 8;
            const int t = idx & 255;
            const float p = sc[s * 260 + t] *
                delta_pos[(((size_t)b * NS + s0 + s) * NS + t) * 3 + x];
            pxh[s * NS + t] = (_Float16)p;
        }
        __syncthreads();

        // attn_x tile (64 x 32padded) = Px (64x256) @ vT-col-tile (256x32)
        v8f c = {};
        for (int kt = 0; kt < 8; ++kt) {
            Frag af, vb;
            #pragma unroll
            for (int j = 0; j < 8; ++j) {
                const int toff = hf * 16 + 2 * j + kt * 32;
                af.u[j] = px[(mt2 * 16 + n16) * 128 + (toff >> 1)];
            }
            const int dcol = nt2 * 16 + n16;
            #pragma unroll
            for (int j = 0; j < 8; ++j) {
                const int t = kt * 32 + hf * 16 + 2 * j;
                vb.u[j] = *(const uint32_t*)(vTbuf + (((size_t)bh * DP) + dcol) * NS + t);
            }
            c = __builtin_amdgcn_wmma_f32_16x16x32_f16(false, af.v, false, vb.v,
                                                       (short)0, c, false, false);
        }
        __syncthreads();   // all px reads done; red may overwrite

        // multiply by Wf slice (guard pad d>=24) and drop into LDS
        const float* wfp = (x == 0) ? Wfx : (x == 1) ? Wfy : Wfz;
        const int d = nt2 * 16 + n16;
        const float wf = (d < ND) ? wfp[h * ND + d] : 0.f;
        #pragma unroll
        for (int i = 0; i < 8; ++i) {
            const int m = i + hf * 8;
            red[(mt2 * 16 + m) * 32 + d] = c[i] * wf;
        }
        __syncthreads();

        if (tid < SB) {
            float s = 0.f;
            #pragma unroll
            for (int dd = 0; dd < 32; ++dd) s += red[tid * 32 + dd];
            partial[(((size_t)b * NS + s0 + tid) * 3 + x) * NH + h] = s;
        }
        __syncthreads();
    }
}

// ---------------------------------------------------------------------------
// Kernel C: reduce partials over heads -> out (b,s,3) f32
// ---------------------------------------------------------------------------
__global__ __launch_bounds__(256) void head_reduce_kernel(
    const float* __restrict__ partial, float* __restrict__ out)
{
    const int idx = blockIdx.x * 256 + threadIdx.x;   // (b*S+s)*3 + x
    if (idx < NB * NS * 3) {
        const float* p = partial + (size_t)idx * NH;
        float s = 0.f;
        #pragma unroll
        for (int i = 0; i < NH; ++i) s += p[i];
        out[idx] = s;
    }
}

extern "C" void kernel_launch(void* const* d_in, const int* in_sizes, int n_in,
                              void* d_out, int out_size, void* d_ws, size_t ws_size,
                              hipStream_t stream) {
    const float* feats     = (const float*)d_in[0];
    const float* attn_bias = (const float*)d_in[1];
    const float* delta_pos = (const float*)d_in[2];
    const float* Wq        = (const float*)d_in[3];
    const float* bq        = (const float*)d_in[4];
    const float* Wk        = (const float*)d_in[5];
    const float* bk        = (const float*)d_in[6];
    const float* Wv        = (const float*)d_in[7];
    const float* bv        = (const float*)d_in[8];
    const float* Wfx       = (const float*)d_in[9];
    const float* Wfy       = (const float*)d_in[10];
    const float* Wfz       = (const float*)d_in[11];

    char* ws = (char*)d_ws;
    const size_t qkv_bytes = (size_t)NB * NH * NS * DP * sizeof(_Float16); // 16 MiB each
    _Float16* qbuf  = (_Float16*)(ws);
    _Float16* kbuf  = (_Float16*)(ws + qkv_bytes);
    _Float16* vTbuf = (_Float16*)(ws + 2 * qkv_bytes);
    float*    part  = (float*)   (ws + 3 * qkv_bytes); // [B,S,3,H] = 3 MiB

    // zero q/k/vT (covers the d>=24 padding lanes)
    hipMemsetAsync(ws, 0, 3 * qkv_bytes, stream);

    // A: 6 col-blocks x 128 row-blocks x 3 matrices, 64x128 tile per block
    qkv_proj_kernel<<<dim3(NE / 128, (NB * NS) / 64, 3), 256, 0, stream>>>(
        feats, Wq, bq, Wk, bk, Wv, bv, qbuf, kbuf, vTbuf);
    // B: (b*h, s-block)
    attn_kernel<<<dim3(NB * NH, NS / SB), 256, 0, stream>>>(
        attn_bias, delta_pos, Wfx, Wfy, Wfz, qbuf, kbuf, vTbuf, part);
    // C: 24576 outputs
    head_reduce_kernel<<<(NB * NS * 3 + 255) / 256, 256, 0, stream>>>(part, (float*)d_out);
}
